// DSAttention_32573031973596
// MI455X (gfx1250) — matrix operand order repaired and checked by reading, not compiled, not run on hardware
//
#include <hip/hip_runtime.h>

// Problem constants: B=4, L=2048, S=2048, H=8, E=64
#define B_ 4
#define L_ 2048
#define S_ 2048
#define H_ 8
#define E_ 64
#define NELEM (B_ * L_ * H_ * E_)   // 4,194,304 elements per tensor
#define LDSPAD 72                   // padded row stride (shorts): 64 data + 8 pad
#define NT (S_ / 64)                // 32 S-tiles
#define TILE_SHORTS (64 * E_)       // 4096 bf16 per 64x64 tile

typedef __attribute__((ext_vector_type(16))) __bf16       v16bf;
typedef __attribute__((ext_vector_type(8)))  float        v8f;
typedef __attribute__((ext_vector_type(8)))  unsigned int v8u;
typedef __attribute__((ext_vector_type(4)))  unsigned int v4u;
typedef __attribute__((ext_vector_type(8)))  int          v8i;
typedef __attribute__((ext_vector_type(4)))  int          v4i;

__device__ __forceinline__ unsigned short f32_to_bf16_rne(float f) {
  unsigned int u = __builtin_bit_cast(unsigned int, f);
  u += 0x7FFFu + ((u >> 16) & 1u);   // round-to-nearest-even
  return (unsigned short)(u >> 16);
}

// Convert fp32 [B,L,H,E] -> bf16 [B,H,L,E] (head-major for contiguous tiles).
__global__ __launch_bounds__(256) void cvt_pack_kernel(const float* __restrict__ src,
                                                       unsigned short* __restrict__ dst) {
  unsigned int idx  = blockIdx.x * 256u + threadIdx.x;   // over B*L*H*E
  unsigned int e    = idx & (E_ - 1);
  unsigned int rest = idx >> 6;          // (b*L + l)*H + h
  unsigned int h    = rest & (H_ - 1);
  unsigned int bl   = rest >> 3;         // b*L + l
  unsigned int l    = bl & (L_ - 1);     // L = 2048 = 2^11
  unsigned int b    = bl >> 11;
  unsigned int didx = (((b * H_ + h) * L_ + l) << 6) + e;
  dst[didx] = f32_to_bf16_rne(src[idx]);
}

// ---- TDM: one descriptor moves a 64x64 bf16 tile global->LDS, with LDS row
// padding (128B data + 16B pad per row) producing the LDSPAD=72 layout. ----
__device__ __forceinline__ void tdm_load_tile(const unsigned short* __restrict__ g,
                                              unsigned short* ldsdst) {
#if __has_builtin(__builtin_amdgcn_tensor_load_to_lds)
  unsigned long long ga = (unsigned long long)(size_t)g;
  unsigned int la = (unsigned int)(size_t)ldsdst;
  // D# group 0: count=1 | lds_addr | global_addr | type=2
  v4u g0 = { 1u, la, (unsigned int)ga,
             (unsigned int)((ga >> 32) & 0x01FFFFFFull) | 0x80000000u };
  // D# group 1: data_size=2B(code1)@16, pad_enable@20, pad_interval=32dw(code4)@22,
  //             pad_amount=4dw(code3)@25; tensor_dim0=64, tensor_dim1=64,
  //             tile_dim0=64, tile_dim1=64, tensor_dim0_stride=64.
  v8i g1 = { (int)((1u << 16) | (1u << 20) | (4u << 22) | (3u << 25)),
             (int)(64u << 16),   // w1[31:16] = tensor_dim0[15:0]
             (int)(64u << 16),   // w2[31:16] = tensor_dim1[15:0]
             (int)(64u << 16),   // w3[31:16] = tile_dim0
             (int)64,            // w4[15:0]  = tile_dim1
             (int)64,            // w5        = tensor_dim0_stride[31:0]
             0, 0 };
  v4i gz = { 0, 0, 0, 0 };       // groups 2/3 unused (2D tensor)
#if defined(__clang_major__) && (__clang_major__ >= 23)
  v8i gz8 = { 0, 0, 0, 0, 0, 0, 0, 0 };
  __builtin_amdgcn_tensor_load_to_lds(g0, g1, gz, gz, gz8, 0);
#else
  __builtin_amdgcn_tensor_load_to_lds(g0, g1, gz, gz, 0);
#endif
#else
  // Fallback: per-lane async copies (ASYNCcnt path)
  int tid = threadIdx.x;
#pragma unroll
  for (int jj = 0; jj < 2; ++jj) {
    int i   = tid + 256 * jj;
    int row = i >> 3;
    int e0  = (i & 7) * 8;
    unsigned int lds_addr = (unsigned int)(size_t)(ldsdst + row * LDSPAD + e0);
    const unsigned short* src = g + i * 8;
    asm volatile("global_load_async_to_lds_b128 %0, %1, off"
                 :: "v"(lds_addr), "v"(src) : "memory");
  }
#endif
}

__device__ __forceinline__ void tdm_wait0() {
#if __has_builtin(__builtin_amdgcn_s_wait_tensorcnt)
  __builtin_amdgcn_s_wait_tensorcnt(0);
#endif
}

// Async-copy the 128x64 bf16 Q tile (16 KB) into LDS (ASYNCcnt path).
__device__ __forceinline__ void stage_q_async(const unsigned short* __restrict__ g,
                                              unsigned short* qs, int tid) {
#pragma unroll
  for (int jj = 0; jj < 4; ++jj) {
    int i   = tid + 256 * jj;       // 1024 chunks of 16B
    int row = i >> 3;               // 8 chunks per 64-e row
    int e0  = (i & 7) * 8;
    unsigned int lds_addr = (unsigned int)(size_t)(qs + row * LDSPAD + e0);
    const unsigned short* src = g + i * 8;
    asm volatile("global_load_async_to_lds_b128 %0, %1, off"
                 :: "v"(lds_addr), "v"(src) : "memory");
  }
}

// Stage one 64x64 bf16 V tile transposed into LDS [e][s] (b128 loads, b16 stores).
__device__ __forceinline__ void stage_v_tr(const unsigned short* __restrict__ g,
                                           unsigned short* vt, int tid) {
#pragma unroll
  for (int jj = 0; jj < 2; ++jj) {
    int i   = tid + 256 * jj;
    int row = i >> 3;
    int e0  = (i & 7) * 8;
    uint4 d = *(const uint4*)(g + i * 8);   // 8 bf16: e0..e0+7 of row
    unsigned int w[4] = {d.x, d.y, d.z, d.w};
#pragma unroll
    for (int q = 0; q < 4; ++q) {
      vt[(e0 + 2 * q    ) * LDSPAD + row] = (unsigned short)(w[q] & 0xFFFFu);
      vt[(e0 + 2 * q + 1) * LDSPAD + row] = (unsigned short)(w[q] >> 16);
    }
  }
}

// Flash-attention: one block = one (b,h) x 128 query rows. 8 waves x 16 rows.
// K tiles via TDM (TENSORcnt), Q via async-to-LDS (ASYNCcnt), V^T manual.
// Double-buffered tiles, one barrier per S-tile.
__global__ __launch_bounds__(256) void attn_kernel(
    const unsigned short* __restrict__ Qb,   // [B,H,L,E] bf16
    const unsigned short* __restrict__ Kb,   // [B,H,S,E] bf16
    const unsigned short* __restrict__ Vb,   // [B,H,S,E] bf16
    const float* __restrict__ tau,           // [B]
    const float* __restrict__ delta,         // [B]
    float* __restrict__ out) {               // [B,L,H,E] f32
  __shared__ __align__(16) unsigned short ktile[2][64 * LDSPAD];  // K tiles [s][e]
  __shared__ __align__(16) unsigned short vtile[2][64 * LDSPAD];  // V tiles [e][s]
  __shared__ __align__(16) unsigned short ptile[8 * 16 * LDSPAD]; // Q stage, then P scratch

  const int tid  = threadIdx.x;
  const int wave = tid >> 5;
  const int lane = tid & 31;
  const int n    = lane & 15;   // N (or A-row) index within 16
  const int kh   = lane >> 4;   // half-wave selector
  const int bh   = blockIdx.y;  // b*H + h
  const int b    = bh >> 3;
  const int h    = bh & 7;
  const int qblk = blockIdx.x;

  const float scale = 0.125f;                    // 1/sqrt(64)
  const float a = tau[b] * scale;                // folded per-batch scale
  const float c = delta[b] * scale;              // folded per-batch bias
  const float LOG2E = 1.4426950408889634f;

  const unsigned short* ksrc = Kb + (size_t)bh * S_ * E_;
  const unsigned short* vsrc = Vb + (size_t)bh * S_ * E_;
  const unsigned short* qsrc = Qb + ((size_t)bh * L_ + qblk * 128) * E_;
  unsigned short* pbase = ptile + wave * 16 * LDSPAD;

  // ---- Prologue: async Q tile, TDM K tile 0, manual V^T tile 0 ----
  stage_q_async(qsrc, ptile, tid);
  if (wave == 0) tdm_load_tile(ksrc, ktile[0]);
  stage_v_tr(vsrc, vtile[0], tid);
  asm volatile("s_wait_asynccnt 0x0" ::: "memory");
  if (wave == 0) tdm_wait0();
  __syncthreads();

  // ---- Q fragments from LDS (ISA bf16 A-layout 16x32, two K-steps for E=64).
  // Wave w's Q rows sit exactly in wave w's private P region: consumed (in-order
  // LDS) before this wave's first P store; other waves never touch this region.
  v8u qa[2];
#pragma unroll
  for (int s = 0; s < 2; ++s)
#pragma unroll
    for (int j = 0; j < 8; ++j) {
      int k = 32 * s + 2 * j + ((j >= 4) ? 8 : 0) + 8 * kh;  // contiguous bf16 pair
      qa[s][j] = *(const unsigned int*)(&ptile[(wave * 16 + n) * LDSPAD + k]);
    }

  v8f o[4];
#pragma unroll
  for (int t = 0; t < 4; ++t) o[t] = {};
  float row_max[8], row_sum[8];
#pragma unroll
  for (int r = 0; r < 8; ++r) { row_max[r] = -3.0e38f; row_sum[r] = 0.0f; }

  for (int st = 0; st < NT; ++st) {
    const int cur = st & 1;
    const unsigned short* kt = ktile[cur];
    const unsigned short* vt = vtile[cur];

    // ---- Kick off next tile into the other buffer (overlaps with compute) ----
    if (st + 1 < NT) {
      if (wave == 0) tdm_load_tile(ksrc + (st + 1) * TILE_SHORTS, ktile[cur ^ 1]);
      stage_v_tr(vsrc + (st + 1) * TILE_SHORTS, vtile[cur ^ 1], tid);
      if (st + 2 < NT) {   // L2 prefetch two tiles ahead (global_prefetch_b8)
        __builtin_prefetch(ksrc + (st + 2) * TILE_SHORTS + tid * 32, 0, 1);
        __builtin_prefetch(vsrc + (st + 2) * TILE_SHORTS + tid * 32, 0, 1);
      }
    }

    // ---- scores = Q x K^T : 4 N-tiles x 2 K-steps of bf16 WMMA ----
    v8f acc[4];
#pragma unroll
    for (int t = 0; t < 4; ++t) {
      acc[t] = {};
#pragma unroll
      for (int s = 0; s < 2; ++s) {
        v8u kf;                            // bf16 B-layout 32x16: B[k][n]=K[16t+n][32s+k]
#pragma unroll
        for (int j = 0; j < 8; ++j)
          kf[j] = *(const unsigned int*)(&kt[(16 * t + n) * LDSPAD + 32 * s + 2 * j + ((j >= 4) ? 8 : 0) + 8 * kh]);
        acc[t] = __builtin_amdgcn_wmma_f32_16x16x32_bf16(
            false, __builtin_bit_cast(v16bf, qa[s]),
            false, __builtin_bit_cast(v16bf, kf),
            (short)0, acc[t], false, false);
      }
    }

    // ---- online softmax (row = r + 8*kh in C/D layout) ----
#pragma unroll
    for (int t = 0; t < 4; ++t)
#pragma unroll
      for (int r = 0; r < 8; ++r)
        acc[t][r] = acc[t][r] * a + c;     // tau/delta/scale folded

#pragma unroll
    for (int r = 0; r < 8; ++r) {
      float tm = fmaxf(fmaxf(acc[0][r], acc[1][r]), fmaxf(acc[2][r], acc[3][r]));
#pragma unroll
      for (int m = 8; m >= 1; m >>= 1)
        tm = fmaxf(tm, __shfl_xor(tm, m, 32));   // reduce across 16-lane N group
      float nm   = fmaxf(row_max[r], tm);
      float corr = exp2f((row_max[r] - nm) * LOG2E);
      row_max[r] = nm;
      float ts = 0.0f;
#pragma unroll
      for (int t = 0; t < 4; ++t) {
        float p = exp2f((acc[t][r] - nm) * LOG2E);
        ts += p;
        pbase[(r + 8 * kh) * LDSPAD + 16 * t + n] = f32_to_bf16_rne(p);
      }
#pragma unroll
      for (int m = 8; m >= 1; m >>= 1)
        ts += __shfl_xor(ts, m, 32);
      row_sum[r] = row_sum[r] * corr + ts;
#pragma unroll
      for (int t = 0; t < 4; ++t) o[t][r] *= corr;
    }

    // ---- O += P x V : reload P in A-layout from LDS, V^T supplies B-layout ----
#pragma unroll
    for (int s = 0; s < 2; ++s) {
      v8u pa;
#pragma unroll
      for (int j = 0; j < 8; ++j) {
        int k = 32 * s + 2 * j + ((j >= 4) ? 8 : 0) + 8 * kh;
        pa[j] = *(const unsigned int*)(&pbase[n * LDSPAD + k]);
      }
#pragma unroll
      for (int t = 0; t < 4; ++t) {
        v8u vf;                            // B[k][n] = V[32s+k][16t+n] = V^T[16t+n][32s+k]
#pragma unroll
        for (int j = 0; j < 8; ++j)
          vf[j] = *(const unsigned int*)(&vt[(16 * t + n) * LDSPAD + 32 * s + 2 * j + ((j >= 4) ? 8 : 0) + 8 * kh]);
        o[t] = __builtin_amdgcn_wmma_f32_16x16x32_bf16(
            false, __builtin_bit_cast(v16bf, pa),
            false, __builtin_bit_cast(v16bf, vf),
            (short)0, o[t], false, false);
      }
    }

    // Next tile's TDM/async copies must have landed before buffers swap.
    asm volatile("s_wait_asynccnt 0x0" ::: "memory");
    if (wave == 0) tdm_wait0();
    __syncthreads();
  }

  // ---- epilogue: normalize and scatter to [B,L,H,E] fp32 ----
#pragma unroll
  for (int r = 0; r < 8; ++r) {
    float inv = 1.0f / row_sum[r];
    int row = qblk * 128 + wave * 16 + r + 8 * kh;
#pragma unroll
    for (int t = 0; t < 4; ++t) {
      int e = 16 * t + n;
      out[(((size_t)b * L_ + row) * H_ + h) * E_ + e] = o[t][r] * inv;
    }
  }
}

extern "C" void kernel_launch(void* const* d_in, const int* in_sizes, int n_in,
                              void* d_out, int out_size, void* d_ws, size_t ws_size,
                              hipStream_t stream) {
  (void)in_sizes; (void)n_in; (void)out_size; (void)ws_size;
  const float* q     = (const float*)d_in[0];
  const float* k     = (const float*)d_in[1];
  const float* v     = (const float*)d_in[2];
  // d_in[3] = attn_mask (unused, identity at eval)
  const float* tau   = (const float*)d_in[4];
  const float* delta = (const float*)d_in[5];
  float* out = (float*)d_out;

  unsigned short* qws = (unsigned short*)d_ws;          // 3 x 8 MB bf16 in workspace
  unsigned short* kws = qws + NELEM;
  unsigned short* vws = kws + NELEM;

  dim3 cgrid(NELEM / 256), cblk(256);
  cvt_pack_kernel<<<cgrid, cblk, 0, stream>>>(q, qws);
  cvt_pack_kernel<<<cgrid, cblk, 0, stream>>>(k, kws);
  cvt_pack_kernel<<<cgrid, cblk, 0, stream>>>(v, vws);

  dim3 agrid(L_ / 128, B_ * H_), ablk(256);
  attn_kernel<<<agrid, ablk, 0, stream>>>(qws, kws, vws, tau, delta, out);
}
